// SelectiveSSM_76811195122365
// MI455X (gfx1250) — matrix operand order repaired and checked by reading, not compiled, not run on hardware
//
#include <hip/hip_runtime.h>
#include <hip/hip_bf16.h>

// ---------------------------------------------------------------------------
// Mamba selective-SSM forward for MI455X (gfx1250, wave32).
// GEMMs: error-compensated bf16 WMMA (v_wmma_f32_16x16x32_bf16, 3 wmma / K=32
// slab: hi*hi + lo*hi + hi*lo) -> fp32-level accuracy at bf16 matrix rate.
// Block tile 128x64, wave tile 32x32 (4 accumulators, 12 WMMA per K-slab).
// Scan: 1 lane per (b,d) channel, 16 SSM states in VGPRs, v_exp_f32 TRANS ops.
// ---------------------------------------------------------------------------

typedef __attribute__((ext_vector_type(16))) __bf16 v16bf;
typedef __attribute__((ext_vector_type(8)))  __bf16 v8bf;
typedef __attribute__((ext_vector_type(4)))  __bf16 v4bf;
typedef __attribute__((ext_vector_type(8)))  float  v8f;

#define BM 128
#define BN 64
#define BK 32

__device__ __forceinline__ float sigmoidf_fast(float x) {
  return 1.0f / (1.0f + __expf(-x));
}
// stable softplus on the fast TRANS path (v_exp_f32 + v_log_f32), branchless
__device__ __forceinline__ float softplusf_fast(float x) {
  float r = __logf(1.0f + __expf(x));
  return (x > 20.0f) ? x : r;
}

// A-fragment (16x32 bf16 tile, ISA 7.12.2 layout): lane L holds row M=L%16;
// K chunks {kh*8..kh*8+7} and {16+kh*8..+7} where kh = L/16.
__device__ __forceinline__ v16bf ld_frag_a(const __bf16* base, int kh) {
  v8bf c0 = *(const v8bf*)(base + kh * 8);
  v8bf c1 = *(const v8bf*)(base + 16 + kh * 8);
  return __builtin_shufflevector(c0, c1, 0, 1, 2, 3, 4, 5, 6, 7,
                                 8, 9, 10, 11, 12, 13, 14, 15);
}

// split one float4 into bf16 hi/lo planes, packed for 8-byte LDS stores
__device__ __forceinline__ void split4(float4 v, v4bf* hi, v4bf* lo) {
  v4bf h, l;
  h[0] = (__bf16)v.x; l[0] = (__bf16)(v.x - (float)h[0]);
  h[1] = (__bf16)v.y; l[1] = (__bf16)(v.y - (float)h[1]);
  h[2] = (__bf16)v.z; l[2] = (__bf16)(v.z - (float)h[2]);
  h[3] = (__bf16)v.w; l[3] = (__bf16)(v.w - (float)h[3]);
  *hi = h; *lo = l;
}

// ---------------------------------------------------------------------------
// GEMM: C[M,N] = A[M,K] @ Bm[N,K]^T  (all row-major), optional epilogue:
//   epi==0 : plain store;  epi==1 : C = softplus(acc + bias[n])
// 256 threads (8 waves). LDS: bf16 hi/lo planes of A(128x32) and B(64x32),
// 24 KB. Wave (wid&3) -> M offset 32*(wid&3); (wid>>2) -> N offset 32*(wid>>2).
// ---------------------------------------------------------------------------
__global__ __launch_bounds__(256)
void gemm_bf16x3(const float* __restrict__ A, const float* __restrict__ Bm,
                 float* __restrict__ C, int M, int N, int K,
                 const float* __restrict__ bias, int epi) {
  __shared__ __align__(32) __bf16 sAh[BM * BK];
  __shared__ __align__(32) __bf16 sAl[BM * BK];
  __shared__ __align__(32) __bf16 sBh[BN * BK];
  __shared__ __align__(32) __bf16 sBl[BN * BK];

  const int tid  = threadIdx.x;
  const int nb   = blockIdx.x;   // N tile
  const int mb   = blockIdx.y;   // M tile
  const int wid  = tid >> 5;
  const int lane = tid & 31;
  const int kh   = lane >> 4;    // K-half this lane holds
  const int r    = lane & 15;
  const int mW   = (wid & 3) * 32;   // wave M offset (2 x 16-row frags)
  const int nW   = (wid >> 2) * 32;  // wave N offset (2 x 16-col frags)

  v8f acc00 = {}, acc01 = {}, acc10 = {}, acc11 = {};

  for (int k0 = 0; k0 < K; k0 += BK) {
    // ---- stage tiles: float4 global loads -> bf16 hi/lo, 8B ds stores ----
#pragma unroll
    for (int i = 0; i < 4; ++i) {              // A: 128x32 = 1024 float4
      int idx4 = tid + i * 256;
      int rr = idx4 >> 3, c = (idx4 & 7) * 4;
      float4 v = *(const float4*)&A[(size_t)(mb * BM + rr) * K + k0 + c];
      v4bf h, l; split4(v, &h, &l);
      *(v4bf*)&sAh[rr * BK + c] = h;
      *(v4bf*)&sAl[rr * BK + c] = l;
    }
#pragma unroll
    for (int i = 0; i < 2; ++i) {              // B: 64x32 = 512 float4
      int idx4 = tid + i * 256;
      int rr = idx4 >> 3, c = (idx4 & 7) * 4;
      float4 v = *(const float4*)&Bm[(size_t)(nb * BN + rr) * K + k0 + c];
      v4bf h, l; split4(v, &h, &l);
      *(v4bf*)&sBh[rr * BK + c] = h;
      *(v4bf*)&sBl[rr * BK + c] = l;
    }
    __syncthreads();

    // ---- fragments ----
    v16bf a0h = ld_frag_a(&sAh[(mW + r) * BK], kh);
    v16bf a0l = ld_frag_a(&sAl[(mW + r) * BK], kh);
    v16bf a1h = ld_frag_a(&sAh[(mW + 16 + r) * BK], kh);
    v16bf a1l = ld_frag_a(&sAl[(mW + 16 + r) * BK], kh);
    // B: lane L holds col N=L%16, K chunks {kh*8..+7} and {16+kh*8..+7}
    v16bf b0h = ld_frag_a(&sBh[(nW + r) * BK], kh);
    v16bf b0l = ld_frag_a(&sBl[(nW + r) * BK], kh);
    v16bf b1h = ld_frag_a(&sBh[(nW + 16 + r) * BK], kh);
    v16bf b1l = ld_frag_a(&sBl[(nW + 16 + r) * BK], kh);

    // ---- 12 WMMA: hi*hi + lo*hi + hi*lo per accumulator ----
#define WMMA(ACC, AV, BV) \
    ACC = __builtin_amdgcn_wmma_f32_16x16x32_bf16(false, AV, false, BV, \
                                                  (short)0, ACC, false, false)
    WMMA(acc00, a0h, b0h); WMMA(acc00, a0l, b0h); WMMA(acc00, a0h, b0l);
    WMMA(acc01, a0h, b1h); WMMA(acc01, a0l, b1h); WMMA(acc01, a0h, b1l);
    WMMA(acc10, a1h, b0h); WMMA(acc10, a1l, b0h); WMMA(acc10, a1h, b0l);
    WMMA(acc11, a1h, b1h); WMMA(acc11, a1l, b1h); WMMA(acc11, a1h, b1l);
#undef WMMA
    __syncthreads();
  }

  // ---- epilogue: C/D layout VGPR i -> row M = kh*8 + i, col N = lane%16 ----
#pragma unroll
  for (int i = 0; i < 8; ++i) {
    int rloc = kh * 8 + i;
    size_t row0 = (size_t)(mb * BM + mW + rloc);
    size_t row1 = (size_t)(mb * BM + mW + 16 + rloc);
    int col0 = nb * BN + nW + r;
    int col1 = col0 + 16;
    float v00 = acc00[i], v01 = acc01[i], v10 = acc10[i], v11 = acc11[i];
    if (epi == 1) {
      float bi0 = bias[col0], bi1 = bias[col1];
      v00 = softplusf_fast(v00 + bi0);
      v01 = softplusf_fast(v01 + bi1);
      v10 = softplusf_fast(v10 + bi0);
      v11 = softplusf_fast(v11 + bi1);
    }
    C[row0 * N + col0] = v00;
    C[row0 * N + col1] = v01;
    C[row1 * N + col0] = v10;
    C[row1 * N + col1] = v11;
  }
  (void)M;
}

// ---------------------------------------------------------------------------
// Causal depthwise conv (k=4) + bias + SiLU.  xz: (M, 4096), first 2048 cols
// are xp.  xc: (M, 2048).  One thread per output element, stride-1 in d.
// ---------------------------------------------------------------------------
__global__ __launch_bounds__(256)
void conv_silu_kernel(const float* __restrict__ xz, const float* __restrict__ cw,
                      const float* __restrict__ cb, float* __restrict__ xc) {
  const int T = 2048;
  int idx = blockIdx.x * 256 + threadIdx.x;   // = m*2048 + d
  int d = idx & 2047;
  int m = idx >> 11;
  int t = m & (T - 1);
  int b = m >> 11;
  float acc = cb[d];
#pragma unroll
  for (int j = 0; j < 4; ++j) {
    int tt = t - 3 + j;
    if (tt >= 0)
      acc += cw[d * 4 + j] * xz[(size_t)(b * T + tt) * 4096 + d];
  }
  xc[idx] = acc * sigmoidf_fast(acc);
}

// ---------------------------------------------------------------------------
// ssm[m, s] = sum_k xc[m,k] * Wx[s,k], s in [0,32). One block per row m.
// ---------------------------------------------------------------------------
__global__ __launch_bounds__(256)
void ssm_proj_kernel(const float* __restrict__ xc, const float* __restrict__ Wx,
                     float* __restrict__ ssm) {
  __shared__ float red[32 * 256];
  const int m = blockIdx.x, tid = threadIdx.x;
  float acc[32];
#pragma unroll
  for (int s = 0; s < 32; ++s) acc[s] = 0.0f;
#pragma unroll
  for (int j = 0; j < 8; ++j) {
    int k = tid + j * 256;
    float xv = xc[(size_t)m * 2048 + k];
#pragma unroll
    for (int s = 0; s < 32; ++s) acc[s] += xv * Wx[s * 2048 + k];
  }
#pragma unroll
  for (int s = 0; s < 32; ++s) red[s * 256 + tid] = acc[s];
  __syncthreads();
  for (int st = 128; st > 0; st >>= 1) {
    if (tid < st) {
#pragma unroll
      for (int s = 0; s < 32; ++s) red[s * 256 + tid] += red[s * 256 + tid + st];
    }
    __syncthreads();
  }
  if (tid < 32) ssm[(size_t)m * 32 + tid] = red[tid * 256];
}

// ---------------------------------------------------------------------------
// Selective scan: one lane per (b,d) channel, 16 states in VGPRs, sequential
// over T. B/C (32 floats per (b,t)) broadcast across the block via cache.
// ---------------------------------------------------------------------------
__global__ __launch_bounds__(256)
void scan_kernel(const float* __restrict__ xc, const float* __restrict__ dtb,
                 const float* __restrict__ ssm, const float* __restrict__ A_log,
                 float* __restrict__ yb) {
  const int T = 2048;
  int gid = blockIdx.x * 256 + threadIdx.x;   // over B*D_INNER = 8192
  int b = gid >> 11;
  int d = gid & 2047;
  float As[16];
#pragma unroll
  for (int s = 0; s < 16; ++s) As[s] = -__expf(A_log[s]);
  float h[16];
#pragma unroll
  for (int s = 0; s < 16; ++s) h[s] = 0.0f;

  for (int t = 0; t < T; ++t) {
    size_t base = (size_t)(b * T + t);
    float dtv = dtb[base * 2048 + d];
    float xv  = xc[base * 2048 + d];
    const float* bc = &ssm[base * 32];
    float dx = dtv * xv;
    float y = 0.0f;
#pragma unroll
    for (int s = 0; s < 16; ++s) {
      float dA = __expf(dtv * As[s]);          // v_exp_f32 (TRANS pipe)
      h[s] = dA * h[s] + dx * bc[s];
      y += h[s] * bc[16 + s];
    }
    yb[base * 2048 + d] = y;
  }
}

// ---------------------------------------------------------------------------
// LayerNorm over D_INNER + D-skip + SiLU(z) gate, in-place on yb.
// One block per row m = (b,t).
// ---------------------------------------------------------------------------
__global__ __launch_bounds__(256)
void ln_gate_kernel(float* __restrict__ yb, const float* __restrict__ xc,
                    const float* __restrict__ xz, const float* __restrict__ g,
                    const float* __restrict__ lb, const float* __restrict__ Dp) {
  __shared__ float r1[256];
  __shared__ float r2[256];
  const int m = blockIdx.x, tid = threadIdx.x;
  float s = 0.0f, s2 = 0.0f;
#pragma unroll
  for (int j = 0; j < 8; ++j) {
    int k = tid + j * 256;
    float v = yb[(size_t)m * 2048 + k];
    s += v;
    s2 += v * v;
  }
  r1[tid] = s;
  r2[tid] = s2;
  __syncthreads();
  for (int st = 128; st > 0; st >>= 1) {
    if (tid < st) { r1[tid] += r1[tid + st]; r2[tid] += r2[tid + st]; }
    __syncthreads();
  }
  float mu  = r1[0] * (1.0f / 2048.0f);
  float var = r2[0] * (1.0f / 2048.0f) - mu * mu;
  float rs  = rsqrtf(var + 1e-5f);
#pragma unroll
  for (int j = 0; j < 8; ++j) {
    int k = tid + j * 256;
    size_t i = (size_t)m * 2048 + k;
    float v = (yb[i] - mu) * rs * g[k] + lb[k] + xc[i] * Dp[k];
    float z = xz[(size_t)m * 4096 + 2048 + k];
    yb[i] = v * (z * sigmoidf_fast(z));
  }
}

// ---------------------------------------------------------------------------
extern "C" void kernel_launch(void* const* d_in, const int* in_sizes, int n_in,
                              void* d_out, int out_size, void* d_ws, size_t ws_size,
                              hipStream_t stream) {
  const float* x      = (const float*)d_in[0];
  const float* W_in   = (const float*)d_in[1];
  const float* conv_w = (const float*)d_in[2];
  const float* conv_b = (const float*)d_in[3];
  const float* W_x    = (const float*)d_in[4];
  const float* W_dt   = (const float*)d_in[5];
  const float* b_dt   = (const float*)d_in[6];
  const float* A_log  = (const float*)d_in[7];
  const float* D_par  = (const float*)d_in[8];
  const float* W_out  = (const float*)d_in[9];
  const float* ln_g   = (const float*)d_in[10];
  const float* ln_b   = (const float*)d_in[11];
  float* out = (float*)d_out;

  const int Bb = 4, T = 2048, DI = 2048;
  const int M = Bb * T;                       // 8192

  // workspace carve (fp32): xz 128MB | xc 64MB | dt 64MB | ssm 1MB | y 64MB
  float* ws  = (float*)d_ws;
  float* xz  = ws;
  float* xc  = xz + (size_t)M * 4096;
  float* dtb = xc + (size_t)M * 2048;
  float* ssm = dtb + (size_t)M * 2048;
  float* yb  = ssm + (size_t)M * 32;

  // 1) xz = x @ W_in^T           (8192 x 1024) x (4096 x 1024)^T
  gemm_bf16x3<<<dim3(4096 / BN, M / BM), 256, 0, stream>>>(
      x, W_in, xz, M, 4096, 1024, nullptr, 0);

  // 2) depthwise causal conv + SiLU -> xc
  conv_silu_kernel<<<(M * DI) / 256, 256, 0, stream>>>(xz, conv_w, conv_b, xc);

  // 3) dt = softplus(xc @ W_dt^T + b_dt)
  gemm_bf16x3<<<dim3(2048 / BN, M / BM), 256, 0, stream>>>(
      xc, W_dt, dtb, M, 2048, 2048, b_dt, 1);

  // 4) ssm (B,C projections, N=32 skinny)
  ssm_proj_kernel<<<M, 256, 0, stream>>>(xc, W_x, ssm);

  // 5) selective scan (sequential over T)
  scan_kernel<<<(Bb * DI) / 256, 256, 0, stream>>>(xc, dtb, ssm, A_log, yb);

  // 6) LayerNorm + skip + gate (in-place on yb)
  ln_gate_kernel<<<M, 256, 0, stream>>>(yb, xc, xz, ln_g, ln_b, D_par);

  // 7) out = yb @ W_out^T        (8192 x 2048) x (1024 x 2048)^T
  gemm_bf16x3<<<dim3(1024 / BN, M / BM), 256, 0, stream>>>(
      yb, W_out, out, M, 1024, 2048, nullptr, 0);

  (void)in_sizes; (void)n_in; (void)out_size; (void)ws_size;
}
